// GraphRetriever_6854767805056
// MI455X (gfx1250) — matrix operand order repaired and checked by reading, not compiled, not run on hardware
//
#include <hip/hip_runtime.h>
#include <hip/hip_bf16.h>
#include <math.h>

typedef __attribute__((ext_vector_type(16))) __bf16 v16bf;
typedef __attribute__((ext_vector_type(8)))  float  v8f;

#define N_NODES 50000
#define DIM     128
#define NREL    16
#define NEDGE   800000
#define NPATH   256
#define PLEN    4

__device__ __forceinline__ unsigned short f2bf(float f) {
  unsigned int u = __float_as_uint(f);
  u += 0x7FFFu + ((u >> 16) & 1u);          // round-to-nearest-even
  return (unsigned short)(u >> 16);
}
__device__ __forceinline__ float sigm(float x) { return 1.0f / (1.0f + expf(-x)); }

// ---------------- utility kernels ----------------
__global__ void k_zero(float* __restrict__ p, long long n) {
  long long i  = (long long)blockIdx.x * blockDim.x + threadIdx.x;
  long long st = (long long)gridDim.x * blockDim.x;
  for (; i < n; i += st) p[i] = 0.0f;
}

__global__ void k_deg(const int* __restrict__ dst, float* __restrict__ deg, int E) {
  int e = blockIdx.x * 256 + threadIdx.x;
  if (e < E)
    __hip_atomic_fetch_add(&deg[dst[e]], 1.0f, __ATOMIC_RELAXED, __HIP_MEMORY_SCOPE_AGENT);
}

__global__ void k_cast_bf16(const float* __restrict__ s, unsigned short* __restrict__ d, long long n) {
  long long i = (long long)blockIdx.x * 256 + threadIdx.x;
  if (i < n) d[i] = f2bf(s[i]);
}

// Pre-swizzle RGCN weights (both layers, 16 relations + self) into the contiguous-k (k = 16*half + 2v)
// WMMA 32x16 bf16 B-fragment layout: uint = {K=k (lo16), K=k+1 (hi16)} per lane.
// Index: Wsw[(l*17+w)*8192 + ((ks*8+nt)*32+lane)*8 + v]
__global__ void k_wswizzle(const float* __restrict__ W_rel, const float* __restrict__ W_self,
                           unsigned int* __restrict__ Wsw) {
  int tid = blockIdx.x * 256 + threadIdx.x;
  const int total = 2 * 17 * 8192;
  if (tid >= total) return;
  int lw   = tid >> 13;          // (l*17+w)
  int rem  = tid & 8191;
  int l    = lw / 17, w = lw % 17;
  int v    = rem & 7;
  int lane = (rem >> 3) & 31;
  int nt   = (rem >> 8) & 7;
  int ks   = rem >> 11;          // 0..3 (K step of 32)
  int half = lane >> 4;
  int n    = nt * 16 + (lane & 15);
  int k    = ks * 32 + half * 16 + 2 * v;
  const float* src;
  if (w < 16) src = W_rel  + ((((size_t)l * 16 + w) * 128 + k) * 128 + n);
  else        src = W_self + (((size_t)l * 128 + k) * 128 + n);
  unsigned int lo = f2bf(src[0]);
  unsigned int hi = f2bf(src[128]);
  Wsw[tid] = lo | (hi << 16);
}

// ---------------- WMMA GEMM: out[w] = X(bf16) @ W[w](bf16), f32 accumulate --------
// grid = (ceil(N/128), 17); block = 256 (8 wave32). w<16 -> trans[w], w==16 -> selfout
__global__ void __launch_bounds__(256) k_gemm(const unsigned short* __restrict__ Xb,
                                              const unsigned int* __restrict__ Wsw,
                                              float* __restrict__ trans,
                                              float* __restrict__ selfout,
                                              int nrows) {
  __shared__ unsigned int Xs[128 * 68];   // 128 rows x 64 dwords, stride 68 (pad)
  int tid     = threadIdx.x;
  int tileRow = blockIdx.x * 128;
  int w       = blockIdx.y;

  const unsigned int* Xu = (const unsigned int*)Xb;
#pragma unroll
  for (int i = 0; i < 32; ++i) {
    int e = i * 256 + tid;
    int r = e >> 6, c = e & 63;
    int grow = tileRow + r;
    unsigned int val = 0u;
    if (grow < nrows) val = Xu[(size_t)grow * 64 + c];
    Xs[r * 68 + c] = val;
  }
  __syncthreads();

  const unsigned int* wp = Wsw + (size_t)w * 8192;
  int wave = tid >> 5, lane = tid & 31, half = lane >> 4, l16 = lane & 15;
  int arow = wave * 16 + l16;

  v8f acc[8];
#pragma unroll
  for (int nt = 0; nt < 8; ++nt)
#pragma unroll
    for (int j = 0; j < 8; ++j) acc[nt][j] = 0.0f;

#pragma unroll
  for (int ks = 0; ks < 4; ++ks) {
    union { v16bf m; unsigned int u[8]; } fa;
    int koff = ks * 32;
#pragma unroll
    for (int v = 0; v < 8; ++v) {                   // A 16x32 bf16 layout (contiguous k = 16*half + s, matches Wsw)
      int kb = koff + half * 16 + v * 2;
      fa.u[v] = Xs[arow * 68 + (kb >> 1)];
    }
#pragma unroll
    for (int nt = 0; nt < 8; ++nt) {
      union { v16bf m; unsigned int u[8]; } fb;
      const unsigned int* bp = wp + (((size_t)ks * 8 + nt) * 32 + lane) * 8;
#pragma unroll
      for (int v = 0; v < 8; ++v) fb.u[v] = bp[v];  // pre-swizzled, coalesced
      acc[nt] = __builtin_amdgcn_wmma_f32_16x16x32_bf16(
          false, fa.m, false, fb.m, (short)0, acc[nt], false, false);
    }
  }

  float* out = (w < 16) ? (trans + (size_t)w * nrows * 128) : selfout;
#pragma unroll
  for (int nt = 0; nt < 8; ++nt) {
#pragma unroll
    for (int v = 0; v < 8; ++v) {                   // C layout: M = v + half*8
      int row = tileRow + wave * 16 + v + half * 8;
      int col = nt * 16 + l16;
      if (row < nrows) out[(size_t)row * 128 + col] = acc[nt][v];
    }
  }
}

// ---------------- edge gather + segment-sum (atomic, agg is L2-resident) ------
__global__ void __launch_bounds__(256) k_scatter(const int* __restrict__ et,
                                                 const int* __restrict__ src,
                                                 const int* __restrict__ dst,
                                                 const float* __restrict__ trans,
                                                 float* __restrict__ agg, int E) {
  long long t = (long long)blockIdx.x * 256 + threadIdx.x;
  int e = (int)(t >> 7);
  int d = (int)(t & 127);
  if (e < E) {
    float val = trans[((size_t)et[e] * N_NODES + src[e]) * 128 + d];
    __hip_atomic_fetch_add(&agg[(size_t)dst[e] * 128 + d], val,
                           __ATOMIC_RELAXED, __HIP_MEMORY_SCOPE_AGENT);
  }
}

// ---------------- fused bias + agg/deg + ReLU + LayerNorm (+ bf16 recast) -----
__global__ void __launch_bounds__(128) k_fuse(const float* __restrict__ selfout,
                                              const float* __restrict__ agg,
                                              const float* __restrict__ deg,
                                              const float* __restrict__ bsl,
                                              const float* __restrict__ lng,
                                              const float* __restrict__ lnb,
                                              float* __restrict__ xout,
                                              unsigned short* __restrict__ Xb) {
  int n = blockIdx.x, d = threadIdx.x;
  __shared__ float s1[128], s2[128];
  float dg = fmaxf(deg[n], 1.0f);
  float v  = selfout[(size_t)n * 128 + d] + bsl[d] + agg[(size_t)n * 128 + d] / dg;
  v = fmaxf(v, 0.0f);
  s1[d] = v; s2[d] = v * v; __syncthreads();
  for (int s = 64; s > 0; s >>= 1) {
    if (d < s) { s1[d] += s1[d + s]; s2[d] += s2[d + s]; }
    __syncthreads();
  }
  float mean = s1[0] * (1.0f / 128.0f);
  float var  = s2[0] * (1.0f / 128.0f) - mean * mean;
  float o = (v - mean) * rsqrtf(var + 1e-5f) * lng[d] + lnb[d];
  xout[(size_t)n * 128 + d] = o;
  Xb[(size_t)n * 128 + d]   = f2bf(o);
}

// ---------------- path encoder ----------------
__global__ void __launch_bounds__(256) k_seq(const float* __restrict__ enc,
                                             const int* __restrict__ pn,
                                             const int* __restrict__ pr,
                                             const float* __restrict__ rel_emb,
                                             float* __restrict__ seq) {
  int b = blockIdx.x;            // p*4 + t
  int p = b >> 2, t = b & 3;
  int j = threadIdx.x;
  float val;
  if (j < 128) val = enc[(size_t)pn[p * 4 + t] * 128 + j];
  else         val = (t == 0) ? 0.0f : rel_emb[(size_t)pr[p * 3 + (t - 1)] * 128 + (j - 128)];
  seq[(size_t)b * 256 + j] = val;
}

__global__ void k_transpose(const float* __restrict__ s, float* __restrict__ d,
                            int rows, int cols) {
  int i = blockIdx.x * 256 + threadIdx.x;
  if (i < rows * cols) {
    int r = i / cols, c = i % cols;
    d[(size_t)c * rows + r] = s[i];
  }
}

__global__ void __launch_bounds__(512) k_lstm(const float* __restrict__ seq,
                                              const float* __restrict__ WihT,
                                              const float* __restrict__ WhhT,
                                              const float* __restrict__ bias,
                                              float* __restrict__ hout, int rev) {
  int p = blockIdx.x, g = threadIdx.x;
  __shared__ float xt[256], hs[128], cs[128], gates[512];
  if (g < 128) { hs[g] = 0.0f; cs[g] = 0.0f; }
  __syncthreads();
  for (int t = 0; t < 4; ++t) {
    int tt = rev ? (3 - t) : t;
    if (g < 256) xt[g] = seq[((size_t)p * 4 + tt) * 256 + g];
    __syncthreads();
    float acc = bias[g];
    for (int j = 0; j < 256; ++j) acc += xt[j] * WihT[(size_t)j * 512 + g];
    for (int j = 0; j < 128; ++j) acc += hs[j] * WhhT[(size_t)j * 512 + g];
    gates[g] = acc;
    __syncthreads();
    if (g < 128) {                       // PyTorch gate order i,f,g,o
      float ig = sigm(gates[g]);
      float fg = sigm(gates[128 + g]);
      float gg = tanhf(gates[256 + g]);
      float og = sigm(gates[384 + g]);
      float c  = fg * cs[g] + ig * gg;
      cs[g] = c;
      hs[g] = og * tanhf(c);
    }
    __syncthreads();
  }
  if (g < 128) hout[(size_t)p * 128 + g] = hs[g];
}

__global__ void __launch_bounds__(128) k_pathout(const float* __restrict__ hf,
                                                 const float* __restrict__ hb,
                                                 const float* __restrict__ W_op,
                                                 const float* __restrict__ b_op,
                                                 float* __restrict__ pe) {
  int p = blockIdx.x, d = threadIdx.x;
  __shared__ float comb[256];
  comb[d]       = hf[(size_t)p * 128 + d];
  comb[128 + d] = hb[(size_t)p * 128 + d];
  __syncthreads();
  float acc = b_op[d];
  for (int j = 0; j < 256; ++j) acc += comb[j] * W_op[(size_t)d * 256 + j];
  pe[(size_t)p * 128 + d] = fmaxf(acc, 0.0f);
}

__global__ void __launch_bounds__(128) k_score(const float* __restrict__ pe,
                                               const float* __restrict__ q,
                                               const float* __restrict__ W_s1,
                                               const float* __restrict__ b_s1,
                                               const float* __restrict__ W_s2,
                                               const float* __restrict__ b_s2,
                                               float* __restrict__ sim) {
  int p = blockIdx.x, d = threadIdx.x;
  __shared__ float comb[256];
  __shared__ float red[128];
  comb[d]       = pe[(size_t)p * 128 + d];
  comb[128 + d] = q[d];
  __syncthreads();
  float acc = b_s1[d];
  for (int j = 0; j < 256; ++j) acc += comb[j] * W_s1[(size_t)d * 256 + j];
  red[d] = fmaxf(acc, 0.0f) * W_s2[d];
  __syncthreads();
  for (int s = 64; s > 0; s >>= 1) {
    if (d < s) red[d] += red[d + s];
    __syncthreads();
  }
  if (d == 0) sim[p] = sigm(red[0] + b_s2[0]);
}

// ---------------- launcher ----------------
extern "C" void kernel_launch(void* const* d_in, const int* in_sizes, int n_in,
                              void* d_out, int out_size, void* d_ws, size_t ws_size,
                              hipStream_t stream) {
  const float* node_features = (const float*)d_in[0];
  const int*   edge_index    = (const int*)d_in[1];
  const int*   edge_types    = (const int*)d_in[2];
  const float* query         = (const float*)d_in[3];
  const int*   path_nodes    = (const int*)d_in[4];
  const int*   path_rel      = (const int*)d_in[5];
  const float* W_self        = (const float*)d_in[6];
  const float* b_self        = (const float*)d_in[7];
  const float* W_rel         = (const float*)d_in[8];
  const float* ln_g          = (const float*)d_in[9];
  const float* ln_b          = (const float*)d_in[10];
  const float* rel_emb       = (const float*)d_in[11];
  const float* W_ih_f        = (const float*)d_in[12];
  const float* W_hh_f        = (const float*)d_in[13];
  const float* b_f           = (const float*)d_in[14];
  const float* W_ih_b        = (const float*)d_in[15];
  const float* W_hh_b        = (const float*)d_in[16];
  const float* b_b           = (const float*)d_in[17];
  const float* W_op          = (const float*)d_in[18];
  const float* b_op          = (const float*)d_in[19];
  const float* W_s1          = (const float*)d_in[20];
  const float* b_s1          = (const float*)d_in[21];
  const float* W_s2          = (const float*)d_in[22];
  const float* b_s2          = (const float*)d_in[23];

  const int* e_src = edge_index;
  const int* e_dst = edge_index + NEDGE;

  char* ws = (char*)d_ws;
  size_t off = 0;
  auto alloc = [&](size_t bytes) -> char* {
    char* p = ws + off;
    off = (off + bytes + 255) & ~(size_t)255;
    return p;
  };
  float*          deg     = (float*)alloc((size_t)N_NODES * 4);
  unsigned short* Xb      = (unsigned short*)alloc((size_t)N_NODES * 128 * 2);
  unsigned int*   Wsw     = (unsigned int*)alloc((size_t)2 * 17 * 8192 * 4);
  float*          trans   = (float*)alloc((size_t)16 * N_NODES * 128 * 4);
  float*          selfout = (float*)alloc((size_t)N_NODES * 128 * 4);
  float*          agg     = (float*)alloc((size_t)N_NODES * 128 * 4);
  float*          xcur    = (float*)alloc((size_t)N_NODES * 128 * 4);
  float*          seq     = (float*)alloc((size_t)NPATH * PLEN * 256 * 4);
  float*          hf      = (float*)alloc((size_t)NPATH * 128 * 4);
  float*          hb      = (float*)alloc((size_t)NPATH * 128 * 4);
  float*          WihTf   = (float*)alloc((size_t)256 * 512 * 4);
  float*          WihTb   = (float*)alloc((size_t)256 * 512 * 4);
  float*          WhhTf   = (float*)alloc((size_t)128 * 512 * 4);
  float*          WhhTb   = (float*)alloc((size_t)128 * 512 * 4);

  float* out_enc = (float*)d_out;
  float* out_pe  = out_enc + (size_t)N_NODES * 128;
  float* out_sim = out_pe + (size_t)NPATH * 128;

  // degree + casts + weight swizzles
  k_zero<<<1024, 256, 0, stream>>>(deg, (long long)N_NODES);
  k_deg<<<(NEDGE + 255) / 256, 256, 0, stream>>>(e_dst, deg, NEDGE);
  k_cast_bf16<<<(unsigned)(((long long)N_NODES * 128 + 255) / 256), 256, 0, stream>>>(
      node_features, Xb, (long long)N_NODES * 128);
  k_wswizzle<<<(2 * 17 * 8192 + 255) / 256, 256, 0, stream>>>(W_rel, W_self, Wsw);
  k_transpose<<<(512 * 256 + 255) / 256, 256, 0, stream>>>(W_ih_f, WihTf, 512, 256);
  k_transpose<<<(512 * 256 + 255) / 256, 256, 0, stream>>>(W_ih_b, WihTb, 512, 256);
  k_transpose<<<(512 * 128 + 255) / 256, 256, 0, stream>>>(W_hh_f, WhhTf, 512, 128);
  k_transpose<<<(512 * 128 + 255) / 256, 256, 0, stream>>>(W_hh_b, WhhTb, 512, 128);

  // two RGCN layers
  for (int l = 0; l < 2; ++l) {
    k_zero<<<4096, 256, 0, stream>>>(agg, (long long)N_NODES * 128);
    dim3 g((N_NODES + 127) / 128, 17);
    k_gemm<<<g, 256, 0, stream>>>(Xb, Wsw + (size_t)l * 17 * 8192, trans, selfout, N_NODES);
    long long st = (long long)NEDGE * 128;
    k_scatter<<<(unsigned)((st + 255) / 256), 256, 0, stream>>>(edge_types, e_src, e_dst,
                                                                trans, agg, NEDGE);
    float* xout = (l == 0) ? xcur : out_enc;
    k_fuse<<<N_NODES, 128, 0, stream>>>(selfout, agg, deg, b_self + l * 128,
                                        ln_g + l * 128, ln_b + l * 128, xout, Xb);
  }

  // path encoder + scorer
  k_seq<<<NPATH * PLEN, 256, 0, stream>>>(out_enc, path_nodes, path_rel, rel_emb, seq);
  k_lstm<<<NPATH, 512, 0, stream>>>(seq, WihTf, WhhTf, b_f, hf, 0);
  k_lstm<<<NPATH, 512, 0, stream>>>(seq, WihTb, WhhTb, b_b, hb, 1);
  k_pathout<<<NPATH, 128, 0, stream>>>(hf, hb, W_op, b_op, out_pe);
  k_score<<<NPATH, 128, 0, stream>>>(out_pe, query, W_s1, b_s1, W_s2, b_s2, out_sim);

  (void)in_sizes; (void)n_in; (void)out_size; (void)ws_size;
}
